// ConditionedMambaBlock_46342697123954
// MI455X (gfx1250) — matrix operand order repaired and checked by reading, not compiled, not run on hardware
//
#include <hip/hip_runtime.h>
#include <hip/hip_bf16.h>

// ---------------------------------------------------------------------------
// ConditionedMambaBlock on gfx1250 (MI455X, wave32, WMMA + Tensor Data Mover)
//
//  0) convert weights fp32 -> bf16 TRANSPOSED (W^T, rows = output col, K contig)
//  1) LayerNorm -> xn (bf16)
//  2) WMMA GEMM (TDM-staged B tiles): xz = xn @ W_in       [4096 x 2048]
//  3) depthwise causal conv(K=4) + SiLU -> uc (fp32 + bf16)
//  4) WMMA GEMM: x_dbl = uc @ W_x                          [4096 x 64]
//  5) dt (= x_dbl[:, :32]) -> bf16
//  6) WMMA GEMM + softplus(.+b_dt): delta                  [4096 x 1024]
//  7) FiLM params: gamma/beta = cond @ film_w + film_b     [2 x 512]
//  8) selective scan (sequential over N, state in regs) -> y (bf16)
//  9) WMMA GEMM + FiLM epilogue -> d_out                   [4096 x 512]
//
// GEMM staging: wave 0 issues a double-buffered tensor_load_to_lds DMA of a
// 64(n) x 32(k) bf16 tile of W^T per K-step; the TDM pad feature (pad 4 DWORDs
// after every 16 DWORDs) lays rows out at pitch 80 B (16B-aligned,
// bank-conflict-free), so each lane builds its v16bf B fragment with two
// ds_load_b128's.  TENSORcnt tracks completion (s_wait_tensorcnt).
// ---------------------------------------------------------------------------

typedef unsigned short u16;
typedef unsigned int   u32;
typedef unsigned long long u64;
typedef __attribute__((ext_vector_type(16))) __bf16 v16bf;
typedef __attribute__((ext_vector_type(8)))  float  v8f;
typedef __attribute__((ext_vector_type(4)))  u32    su4;
typedef __attribute__((ext_vector_type(8)))  u32    su8;

union ABFrag {
    v16bf v;
    uint4 q[2];
    u32   u[8];
    u16   h[16];
};

#define B_   2
#define N_   2048
#define D_   512
#define E_   1024
#define S_   16
#define R_   32
#define C_   512
#define M_   (B_ * N_)   // 4096

__device__ __forceinline__ u16 f2bf(float f) {
    u32 u = __float_as_uint(f);
    u += 0x7fffu + ((u >> 16) & 1u);   // round-to-nearest-even
    return (u16)(u >> 16);
}

__device__ __forceinline__ float softplusf(float x) {
    return (x > 20.f) ? x : log1pf(__expf(x));
}

// ---------------------------------------------------------------------------
// fp32 [K,N] -> bf16 transposed [N,K]
// ---------------------------------------------------------------------------
__global__ void cvt_bf16_T_kernel(const float* __restrict__ src,
                                  u16* __restrict__ dst, int K, int N) {
    int i = blockIdx.x * blockDim.x + threadIdx.x;
    if (i >= K * N) return;
    int n = i / K, k = i % K;
    dst[i] = f2bf(src[(size_t)k * N + n]);
}

// ---------------------------------------------------------------------------
// LayerNorm over D=512, 256 threads/row, writes bf16
// ---------------------------------------------------------------------------
__global__ __launch_bounds__(256)
void ln_bf16_kernel(const float* __restrict__ x, const float* __restrict__ g,
                    const float* __restrict__ b, u16* __restrict__ out, int Dd) {
    int row = blockIdx.x;
    int t = threadIdx.x;
    const float* xr = x + (size_t)row * Dd;
    float v0 = xr[t], v1 = xr[t + 256];
    __shared__ float s1[256], s2[256];
    s1[t] = v0 + v1;
    s2[t] = v0 * v0 + v1 * v1;
    __syncthreads();
    for (int off = 128; off > 0; off >>= 1) {
        if (t < off) { s1[t] += s1[t + off]; s2[t] += s2[t + off]; }
        __syncthreads();
    }
    float mu  = s1[0] / (float)Dd;
    float var = s2[0] / (float)Dd - mu * mu;
    float rs  = rsqrtf(var + 1e-5f);
    out[(size_t)row * Dd + t]       = f2bf((v0 - mu) * rs * g[t]       + b[t]);
    out[(size_t)row * Dd + t + 256] = f2bf((v1 - mu) * rs * g[t + 256] + b[t + 256]);
}

// ---------------------------------------------------------------------------
// bf16 WMMA GEMM with TDM-staged B tiles.
//   C[M,N] = epilogue(A[M,K] @ W[K,N]),  BwT = W^T as bf16 [N,K] row-major.
// block = 128 threads (4 waves); block tile 64(M) x 64(N); wave = 16 x 64.
// MODE 0: none   MODE 1: softplus(x + bias[n])   MODE 2: gamma*x + beta (FiLM)
// ---------------------------------------------------------------------------
#define LDS_PITCH 40   // u16 units: 32 payload + 4-DWORD TDM pad -> 80 B rows

template <int MODE>
__global__ __launch_bounds__(128)
void wmma_gemm_kernel(const u16* __restrict__ A, const u16* __restrict__ BwT,
                      float* __restrict__ C, int M, int K, int N,
                      const float* __restrict__ bias,
                      const float* __restrict__ gamma,
                      const float* __restrict__ beta, int seqLen) {
    __shared__ u16 ldsB[2][64 * LDS_PITCH];   // double-buffered TDM destination

    const int tid  = threadIdx.x;
    const int wave = tid >> 5;
    const int lane = tid & 31;
    const int half = (lane < 16) ? 0 : 1;
    const int l16  = lane & 15;

    const int m0 = blockIdx.x * 64 + wave * 16;
    const int n0 = blockIdx.y * 64;

    v8f acc[4];
    v8f zero = {0.f, 0.f, 0.f, 0.f, 0.f, 0.f, 0.f, 0.f};
#pragma unroll
    for (int t = 0; t < 4; ++t) acc[t] = zero;

    const u16* Arow = A + (size_t)(m0 + l16) * K;

    // --- TDM issue: 64(n-rows) x 32(k) bf16 tile of W^T into LDS buffer ---
    // D# per CDNA5 ISA.
    // group1 word0: data_size=1(2B)<<16 | pad_en<<20 | pad_interval=3(16dw)<<22
    //               | pad_amount=3(4dw)<<25  -> rows land at pitch 80 B.
    auto tdm_issue = [&](int kb, int buf) {
        u64 ga = (u64)(uintptr_t)BwT + ((u64)(u32)n0 * (u64)(u32)K + (u64)(u32)kb) * 2ull;
        u32 ldsOff = (u32)(uintptr_t)(&ldsB[buf][0]);
        su4 g0;
        g0[0] = 1u;                                   // count=1, user descriptor
        g0[1] = ldsOff;                               // lds_addr (bytes)
        g0[2] = (u32)ga;                              // global_addr[31:0]
        g0[3] = (u32)((ga >> 32) & 0x1FFFFFFu) | 0x80000000u;  // addr[56:32]|type=2
        su8 g1;
        g1[0] = 0x06D10000u;                          // dsize=2B, pad 4dw per 16dw
        g1[1] = ((u32)K & 0xffffu) << 16;             // tensor_dim0[15:0]
        g1[2] = (((u32)K >> 16) & 0xffffu) | (((u32)N & 0xffffu) << 16); // d0 hi|d1 lo
        g1[3] = (((u32)N >> 16) & 0xffffu) | (32u << 16);  // d1 hi | tile_dim0=32
        g1[4] = 64u;                                  // tile_dim1=64, tile_dim2=0
        g1[5] = (u32)K;                               // tensor_dim0_stride lo
        g1[6] = 0u;
        g1[7] = 0u;
        asm volatile("tensor_load_to_lds %0, %1" :: "s"(g0), "s"(g1) : "memory");
    };

    if (wave == 0) tdm_issue(0, 0);
    int cur = 0;

    for (int kb = 0; kb < K; kb += 32) {
        if (wave == 0) {
            if (kb + 32 < K) {
                tdm_issue(kb + 32, cur ^ 1);          // prefetch next tile
                __builtin_amdgcn_s_wait_tensorcnt(1); // current tile done
            } else {
                __builtin_amdgcn_s_wait_tensorcnt(0);
            }
        }
        __syncthreads();   // tile `cur` visible to all waves

        // --- A fragment: 16-bit A 16x32 ISA layout ---
        // lanes 0-15: K 0..7 (v0-3), 16..23 (v4-7); lanes 16-31: 8..15, 24..31
        ABFrag a;
        const u16* ap = Arow + kb + (half ? 8 : 0);
        a.q[0] = *(const uint4*)(ap);
        a.q[1] = *(const uint4*)(ap + 16);

        // --- 4 N-tiles, reuse A fragment; B frag = 2 aligned b128 LDS reads ---
#pragma unroll
        for (int t = 0; t < 4; ++t) {
            ABFrag bfr;
            const u16* bp = &ldsB[cur][(t * 16 + l16) * LDS_PITCH + half * 8];
            bfr.q[0] = *(const uint4*)(bp);
            bfr.q[1] = *(const uint4*)(bp + 16);
            acc[t] = __builtin_amdgcn_wmma_f32_16x16x32_bf16(
                false, a.v, false, bfr.v, (short)0, acc[t], false, false);
        }
        __syncthreads();   // everyone done with `cur` before it is re-filled
        cur ^= 1;
    }

    // --- epilogue: lane<16 -> M=r, lane>=16 -> M=8+r; N=lane&15 ---
    const int mBase = m0 + half * 8;
#pragma unroll
    for (int t = 0; t < 4; ++t) {
        int n = n0 + t * 16 + l16;
#pragma unroll
        for (int r = 0; r < 8; ++r) {
            int m = mBase + r;
            float v = acc[t][r];
            if (MODE == 1) v = softplusf(v + bias[n]);
            if (MODE == 2) {
                int bb = m / seqLen;
                v = gamma[(size_t)bb * N + n] * v + beta[(size_t)bb * N + n];
            }
            C[(size_t)m * N + n] = v;
        }
    }
}

// ---------------------------------------------------------------------------
// depthwise causal conv (K=4) + SiLU; u = xz[:, :E]
// ---------------------------------------------------------------------------
__global__ void conv_silu_kernel(const float* __restrict__ xz,
                                 const float* __restrict__ cw,
                                 const float* __restrict__ cb,
                                 float* __restrict__ uc, u16* __restrict__ ucb,
                                 int total, int Ntok, int Ee) {
    int i = blockIdx.x * blockDim.x + threadIdx.x;
    if (i >= total) return;
    int e  = i % Ee;
    int bn = i / Ee;
    int n  = bn % Ntok;
    float acc = cb[e];
#pragma unroll
    for (int k = 0; k < 4; ++k) {
        int nn = n - 3 + k;
        if (nn >= 0)
            acc += xz[(size_t)(bn - 3 + k) * (2 * Ee) + e] * cw[e * 4 + k];
    }
    float r = acc * (1.f / (1.f + __expf(-acc)));   // SiLU
    uc[i]  = r;
    ucb[i] = f2bf(r);
}

// ---------------------------------------------------------------------------
// dt = x_dbl[:, :32] -> bf16
// ---------------------------------------------------------------------------
__global__ void dt_cvt_kernel(const float* __restrict__ xdbl,
                              u16* __restrict__ dtb, int total) {
    int i = blockIdx.x * blockDim.x + threadIdx.x;
    if (i >= total) return;
    int m = i >> 5, c = i & 31;
    dtb[i] = f2bf(xdbl[(size_t)m * 64 + c]);
}

// ---------------------------------------------------------------------------
// FiLM params: gamma/beta = cond @ film_w + film_b   [B x D]
// ---------------------------------------------------------------------------
__global__ void film_params_kernel(const float* __restrict__ cond,
                                   const float* __restrict__ gw,
                                   const float* __restrict__ gb,
                                   const float* __restrict__ bw,
                                   const float* __restrict__ bb,
                                   float* __restrict__ gamma,
                                   float* __restrict__ beta,
                                   int Bc, int Cdim, int Dd) {
    int i = blockIdx.x * blockDim.x + threadIdx.x;
    if (i >= Bc * Dd) return;
    int b = i / Dd, d = i % Dd;
    const float* cr = cond + (size_t)b * Cdim;
    float ag = 0.f, ab = 0.f;
    for (int c = 0; c < Cdim; ++c) {
        float cv = cr[c];
        ag += cv * gw[(size_t)c * Dd + d];
        ab += cv * bw[(size_t)c * Dd + d];
    }
    gamma[i] = ag + gb[d];
    beta[i]  = ab + bb[d];
}

// ---------------------------------------------------------------------------
// Selective scan: one thread per (b,e), h[S] in registers, Bm/Cm via LDS.
// y = (scan + D_skip*uc) * silu(z) -> bf16 for the final GEMM.
// ---------------------------------------------------------------------------
__global__ __launch_bounds__(256)
void scan_kernel(const float* __restrict__ delta, const float* __restrict__ uc,
                 const float* __restrict__ xz, const float* __restrict__ xdbl,
                 const float* __restrict__ A_log, const float* __restrict__ Dsk,
                 u16* __restrict__ ybf, int Ntok, int Ee) {
    const int blocksPerB = Ee / 256;
    const int b = blockIdx.x / blocksPerB;
    const int e = (blockIdx.x % blocksPerB) * 256 + threadIdx.x;

    float Aes[S_];
#pragma unroll
    for (int s = 0; s < S_; ++s) Aes[s] = -__expf(A_log[(size_t)e * S_ + s]);
    float h[S_];
#pragma unroll
    for (int s = 0; s < S_; ++s) h[s] = 0.f;
    const float dsk = Dsk[e];

    __shared__ float sBC[32];   // Bm[0..15], Cm[0..15]

    for (int n = 0; n < Ntok; ++n) {
        size_t rbn = (size_t)b * Ntok + n;
        if (threadIdx.x < 32) sBC[threadIdx.x] = xdbl[rbn * 64 + 32 + threadIdx.x];
        __syncthreads();

        float dv = delta[rbn * Ee + e];
        float uv = uc[rbn * Ee + e];
        float zv = xz[rbn * (2 * Ee) + Ee + e];
        float du = dv * uv;
        float y = 0.f;
#pragma unroll
        for (int s = 0; s < S_; ++s) {
            float dA = __expf(dv * Aes[s]);
            h[s] = dA * h[s] + du * sBC[s];
            y += h[s] * sBC[16 + s];
        }
        y += dsk * uv;
        y *= zv * (1.f / (1.f + __expf(-zv)));   // * silu(z)
        ybf[rbn * Ee + e] = f2bf(y);

        if (n + 1 < Ntok) {   // hint next step's operands toward the caches
            __builtin_prefetch(&delta[(rbn + 1) * Ee + e], 0, 1);
            __builtin_prefetch(&uc[(rbn + 1) * Ee + e], 0, 1);
            __builtin_prefetch(&xz[(rbn + 1) * (2 * Ee) + Ee + e], 0, 1);
        }
        __syncthreads();
    }
}

// ---------------------------------------------------------------------------
// host launch
// ---------------------------------------------------------------------------
extern "C" void kernel_launch(void* const* d_in, const int* in_sizes, int n_in,
                              void* d_out, int out_size, void* d_ws, size_t ws_size,
                              hipStream_t stream) {
    const float* x       = (const float*)d_in[0];
    const float* cond    = (const float*)d_in[1];
    const float* ln_g    = (const float*)d_in[2];
    const float* ln_b    = (const float*)d_in[3];
    const float* W_in    = (const float*)d_in[4];
    const float* conv_w  = (const float*)d_in[5];
    const float* conv_b  = (const float*)d_in[6];
    const float* W_x     = (const float*)d_in[7];
    const float* W_dt    = (const float*)d_in[8];
    const float* b_dt    = (const float*)d_in[9];
    const float* A_log   = (const float*)d_in[10];
    const float* D_skip  = (const float*)d_in[11];
    const float* W_out   = (const float*)d_in[12];
    const float* film_gw = (const float*)d_in[13];
    const float* film_gb = (const float*)d_in[14];
    const float* film_bw = (const float*)d_in[15];
    const float* film_bb = (const float*)d_in[16];
    float* out = (float*)d_out;

    // workspace carve-up (256B aligned)
    char* wp = (char*)d_ws;
    auto carve = [&](size_t bytes) -> void* {
        void* p = (void*)wp;
        wp += (bytes + 255) & ~(size_t)255;
        return p;
    };
    u16*   W_in_t  = (u16*)carve((size_t)D_ * 2 * E_ * 2);   // [2E, D]
    u16*   W_x_t   = (u16*)carve((size_t)E_ * 64 * 2);       // [64, E]
    u16*   W_dt_t  = (u16*)carve((size_t)R_ * E_ * 2);       // [E, R]
    u16*   W_out_t = (u16*)carve((size_t)E_ * D_ * 2);       // [D, E]
    u16*   xn_b    = (u16*)carve((size_t)M_ * D_ * 2);
    float* xz      = (float*)carve((size_t)M_ * 2 * E_ * 4);
    float* uc      = (float*)carve((size_t)M_ * E_ * 4);
    u16*   uc_b    = (u16*)carve((size_t)M_ * E_ * 2);
    float* x_dbl   = (float*)carve((size_t)M_ * 64 * 4);
    u16*   dt_b    = (u16*)carve((size_t)M_ * R_ * 2);
    float* delta   = (float*)carve((size_t)M_ * E_ * 4);
    u16*   y_b     = (u16*)carve((size_t)M_ * E_ * 2);
    float* gamma   = (float*)carve((size_t)B_ * D_ * 4);
    float* beta    = (float*)carve((size_t)B_ * D_ * 4);

    // 0) weight conversions (fp32 [K,N] -> bf16 [N,K])
    auto cvtT = [&](const float* s, u16* d, int K, int N) {
        cvt_bf16_T_kernel<<<(K * N + 255) / 256, 256, 0, stream>>>(s, d, K, N);
    };
    cvtT(W_in,  W_in_t,  D_, 2 * E_);
    cvtT(W_x,   W_x_t,   E_, 64);
    cvtT(W_dt,  W_dt_t,  R_, E_);
    cvtT(W_out, W_out_t, E_, D_);

    // 1) LayerNorm
    ln_bf16_kernel<<<M_, 256, 0, stream>>>(x, ln_g, ln_b, xn_b, D_);

    // 2) xz = xn @ W_in   [4096 x 2048]
    wmma_gemm_kernel<0><<<dim3(M_ / 64, (2 * E_) / 64), 128, 0, stream>>>(
        xn_b, W_in_t, xz, M_, D_, 2 * E_, nullptr, nullptr, nullptr, N_);

    // 3) conv + SiLU
    {
        int total = M_ * E_;
        conv_silu_kernel<<<(total + 255) / 256, 256, 0, stream>>>(
            xz, conv_w, conv_b, uc, uc_b, total, N_, E_);
    }

    // 4) x_dbl = uc @ W_x   [4096 x 64]
    wmma_gemm_kernel<0><<<dim3(M_ / 64, 64 / 64), 128, 0, stream>>>(
        uc_b, W_x_t, x_dbl, M_, E_, 64, nullptr, nullptr, nullptr, N_);

    // 5) dt -> bf16
    dt_cvt_kernel<<<(M_ * R_ + 255) / 256, 256, 0, stream>>>(x_dbl, dt_b, M_ * R_);

    // 6) delta = softplus(dt @ W_dt + b_dt)   [4096 x 1024]
    wmma_gemm_kernel<1><<<dim3(M_ / 64, E_ / 64), 128, 0, stream>>>(
        dt_b, W_dt_t, delta, M_, R_, E_, b_dt, nullptr, nullptr, N_);

    // 7) FiLM params
    film_params_kernel<<<(B_ * D_ + 255) / 256, 256, 0, stream>>>(
        cond, film_gw, film_gb, film_bw, film_bb, gamma, beta, B_, C_, D_);

    // 8) selective scan -> y (bf16)
    scan_kernel<<<B_ * (E_ / 256), 256, 0, stream>>>(
        delta, uc, xz, x_dbl, A_log, D_skip, y_b, N_, E_);

    // 9) out = gamma * (y @ W_out) + beta   [4096 x 512]
    wmma_gemm_kernel<2><<<dim3(M_ / 64, D_ / 64), 128, 0, stream>>>(
        y_b, W_out_t, out, M_, E_, D_, nullptr, gamma, beta, N_);
}
